// SSS_14654428414310
// MI455X (gfx1250) — matrix operand order repaired and checked by reading, not compiled, not run on hardware
//
#include <hip/hip_runtime.h>
#include <hip/hip_bf16.h>

// ---------------------------------------------------------------------------
// Shapes
// ---------------------------------------------------------------------------
#define BATCH   32
#define CDIM    384
#define NHEAD   8
#define HDIM    48
#define HIDDEN  1536
#define LTOK    1024                  // 32*32
#define MROWS   (BATCH * LTOK)        // 32768 tokens
#define INV_PI  0.31830988618379067f

typedef __attribute__((ext_vector_type(16))) __bf16        bf16x16;
typedef __attribute__((ext_vector_type(8)))  float         f32x8;
typedef __attribute__((ext_vector_type(4)))  unsigned int  u32x4;

union FragAB {
    bf16x16 v;
    u32x4   u[2];
};

// ---------------------------------------------------------------------------
// WMMA fragment loaders (wave32, contiguous k order, same for A and B)
// A (16x32 bf16, row-major [M x K]):
//   lane 0-15: M=lane, elems 0..15 -> K=k0..k0+15
//   lane16-31: M=lane-16, elems 0..15 -> K=k0+16..k0+31
// B (32x16 bf16, from W^T stored [N x K] row-major):
//   lane 0-15: N=lane,    elems 0..15 -> K=k0..k0+15
//   lane16-31: N=lane-16, elems 0..15 -> K=k0+16..k0+31
// ---------------------------------------------------------------------------
__device__ __forceinline__ bf16x16 load_frag_a(const __hip_bfloat16* __restrict__ A,
                                               int lda, int m, int k, int lane) {
    const int row = m + (lane & 15);
    const int kb  = k + (lane & 16);                  // +16 elems for upper half-wave
    const u32x4* p = reinterpret_cast<const u32x4*>(A + (size_t)row * lda + kb);
    FragAB f;
    f.u[0] = p[0];                                    // K = kb .. kb+7
    f.u[1] = p[1];                                    // K = kb+8 .. kb+15
    return f.v;
}

__device__ __forceinline__ bf16x16 load_frag_b(const __hip_bfloat16* __restrict__ Bt,
                                               int ldb, int n, int k, int lane) {
    const int col = n + (lane & 15);
    const int kb  = k + (lane & 16);                  // +16 elems for upper half-wave
    const u32x4* p = reinterpret_cast<const u32x4*>(Bt + (size_t)col * ldb + kb);
    FragAB f;
    f.u[0] = p[0];                                    // K = kb .. kb+7
    f.u[1] = p[1];                                    // K = kb+8 .. kb+15
    return f.v;
}

// ---------------------------------------------------------------------------
// Templated WMMA GEMM:  C[M x N] = A[M x K](bf16) * Bt[N x K]^T(bf16) (+ bias)
// Block: 256 threads = 8 waves; block tile 128(M) x 128(N); wave tile 32 x 64.
// Epilogues: 0 = store f32          (qkv)
//            1 = +bias, SiLU -> bf16 (fc1)
//            2 = +bias -> f32        (fc2)
//            3 = +bias -> f32 scattered to NCHW [B, N, 1024]  (proj)
// ---------------------------------------------------------------------------
template <int EPI>
__global__ __launch_bounds__(256)
void wmma_gemm_kernel(const __hip_bfloat16* __restrict__ A,
                      const __hip_bfloat16* __restrict__ Bt,
                      const float* __restrict__ bias,
                      void* __restrict__ outp,
                      int M, int N, int K)
{
    const int lane = threadIdx.x & 31;
    const int wave = threadIdx.x >> 5;
    const int wm   = wave >> 1;                       // 0..3
    const int wn   = wave & 1;                        // 0..1
    const int m_base = blockIdx.y * 128 + wm * 32;
    const int n_base = blockIdx.x * 128 + wn * 64;

    f32x8 acc[2][4];
    const f32x8 zero = {0.f, 0.f, 0.f, 0.f, 0.f, 0.f, 0.f, 0.f};
    #pragma unroll
    for (int i = 0; i < 2; ++i)
        #pragma unroll
        for (int j = 0; j < 4; ++j)
            acc[i][j] = zero;

    for (int k0 = 0; k0 < K; k0 += 32) {
        const bf16x16 a0 = load_frag_a(A, K, m_base,      k0, lane);
        const bf16x16 a1 = load_frag_a(A, K, m_base + 16, k0, lane);
        const bf16x16 b0 = load_frag_b(Bt, K, n_base,      k0, lane);
        const bf16x16 b1 = load_frag_b(Bt, K, n_base + 16, k0, lane);
        const bf16x16 b2 = load_frag_b(Bt, K, n_base + 32, k0, lane);
        const bf16x16 b3 = load_frag_b(Bt, K, n_base + 48, k0, lane);

        acc[0][0] = __builtin_amdgcn_wmma_f32_16x16x32_bf16(false, a0, false, b0, (short)0, acc[0][0], false, false);
        acc[0][1] = __builtin_amdgcn_wmma_f32_16x16x32_bf16(false, a0, false, b1, (short)0, acc[0][1], false, false);
        acc[0][2] = __builtin_amdgcn_wmma_f32_16x16x32_bf16(false, a0, false, b2, (short)0, acc[0][2], false, false);
        acc[0][3] = __builtin_amdgcn_wmma_f32_16x16x32_bf16(false, a0, false, b3, (short)0, acc[0][3], false, false);
        acc[1][0] = __builtin_amdgcn_wmma_f32_16x16x32_bf16(false, a1, false, b0, (short)0, acc[1][0], false, false);
        acc[1][1] = __builtin_amdgcn_wmma_f32_16x16x32_bf16(false, a1, false, b1, (short)0, acc[1][1], false, false);
        acc[1][2] = __builtin_amdgcn_wmma_f32_16x16x32_bf16(false, a1, false, b2, (short)0, acc[1][2], false, false);
        acc[1][3] = __builtin_amdgcn_wmma_f32_16x16x32_bf16(false, a1, false, b3, (short)0, acc[1][3], false, false);
    }

    // C/D layout: vgpr r -> M = m_tile + r + 8*(lane>=16), N = n_tile + (lane&15)
    const int r_off = (lane & 16) >> 1;               // 0 or 8
    const int cn    = lane & 15;
    #pragma unroll
    for (int i = 0; i < 2; ++i) {
        #pragma unroll
        for (int j = 0; j < 4; ++j) {
            const int n = n_base + j * 16 + cn;
            const float bv = (EPI != 0) ? bias[n] : 0.f;
            #pragma unroll
            for (int r = 0; r < 8; ++r) {
                const int m = m_base + i * 16 + r_off + r;
                float v = acc[i][j][r];
                if (EPI == 0) {
                    ((float*)outp)[(size_t)m * N + n] = v;
                } else if (EPI == 1) {
                    v += bv;
                    v = v / (1.f + __expf(-v));       // SiLU
                    ((__hip_bfloat16*)outp)[(size_t)m * N + n] = __float2bfloat16(v);
                } else if (EPI == 2) {
                    ((float*)outp)[(size_t)m * N + n] = v + bv;
                } else {                              // NCHW scatter: m = b*1024 + l
                    const int bb = m >> 10;
                    const int l  = m & 1023;
                    ((float*)outp)[((size_t)bb * N + n) * 1024 + l] = v + bv;
                }
            }
        }
    }
}

// ---------------------------------------------------------------------------
// Weight convert + transpose:  W[K x N] f32  ->  Wt[N x K] bf16
// ---------------------------------------------------------------------------
__global__ void convert_transpose_kernel(const float* __restrict__ W,
                                         __hip_bfloat16* __restrict__ Wt,
                                         int K, int N)
{
    const int idx = blockIdx.x * 256 + threadIdx.x;
    if (idx >= K * N) return;
    const int n = idx / K;
    const int k = idx - n * K;
    Wt[idx] = __float2bfloat16(W[(size_t)k * N + n]);
}

// ---------------------------------------------------------------------------
// Fused [B,C,H,W] -> [B,L,C] transpose + LayerNorm, emitting bf16 xs and xn.
// One block = 32 consecutive tokens of one batch image; LDS tile transpose.
// ---------------------------------------------------------------------------
#define CPAD 385
__global__ __launch_bounds__(256)
void ln_kernel(const float* __restrict__ x,
               const float* __restrict__ g,
               const float* __restrict__ beta,
               __hip_bfloat16* __restrict__ xs,
               __hip_bfloat16* __restrict__ xn)
{
    __shared__ float tile[32][CPAD];                  // [token][channel]
    const int blk = blockIdx.x;                       // 0..1023
    const int b   = blk >> 5;
    const int l0  = (blk & 31) << 5;
    const int tid = threadIdx.x;

    // coalesced load: for each channel, 32 consecutive l are contiguous
    #pragma unroll
    for (int i = 0; i < 48; ++i) {
        const int e  = i * 256 + tid;                 // e < 384*32
        const int c  = e >> 5;
        const int li = e & 31;
        tile[li][c] = x[(((size_t)b * CDIM + c) << 10) + l0 + li];
    }
    __syncthreads();

    const int lane = tid & 31;
    const int wv   = tid >> 5;
    #pragma unroll
    for (int t4 = 0; t4 < 4; ++t4) {
        const int t = wv * 4 + t4;
        float s = 0.f, ss = 0.f;
        #pragma unroll
        for (int i = 0; i < 12; ++i) {
            const float v = tile[t][lane + (i << 5)];
            s += v; ss += v * v;
        }
        #pragma unroll
        for (int off = 16; off > 0; off >>= 1) {      // wave32 reduction
            s  += __shfl_xor(s,  off, 32);
            ss += __shfl_xor(ss, off, 32);
        }
        const float mu   = s * (1.f / CDIM);
        const float var  = ss * (1.f / CDIM) - mu * mu;
        const float rstd = rsqrtf(var + 1e-5f);
        const size_t row = (((size_t)b) << 10) + l0 + t;
        #pragma unroll
        for (int i = 0; i < 12; ++i) {
            const int c   = lane + (i << 5);
            const float v = tile[t][c];
            xs[row * CDIM + c] = __float2bfloat16(v);
            xn[row * CDIM + c] = __float2bfloat16((v - mu) * rstd * g[c] + beta[c]);
        }
    }
}

// ---------------------------------------------------------------------------
// Linear attention per (b, h):  attn = kn^T v (48x48), o = 0.5 v + (1/pi) qn attn,
// renormalize o, add MLP branch, emit bf16 [M x C] rows for proj GEMM.
// ---------------------------------------------------------------------------
__global__ __launch_bounds__(256)
void attn_kernel(const float* __restrict__ qkv,       // [M x 1152]
                 const float* __restrict__ mid,       // [M x 384]
                 __hip_bfloat16* __restrict__ o)      // [M x 384]
{
    __shared__ float k_s[64][HDIM + 1];
    __shared__ float v_s[64][HDIM + 1];
    __shared__ float attn_s[HDIM][HDIM + 1];
    __shared__ float rnorm[64];

    const int b    = blockIdx.x >> 3;
    const int h    = blockIdx.x & 7;
    const int tid  = threadIdx.x;
    const int hcol = h * HDIM;
    const size_t base = (((size_t)b) << 10) * 1152;

    int dd[9], ee[9];
    float acc[9];
    #pragma unroll
    for (int i = 0; i < 9; ++i) {
        const int p = tid + (i << 8);                 // 0..2303 covers 48x48
        dd[i] = p / HDIM;
        ee[i] = p - dd[i] * HDIM;
        acc[i] = 0.f;
    }

    // ---- pass 1: attn = kn^T v over L in 64-token chunks ----
    for (int lc = 0; lc < LTOK; lc += 64) {
        #pragma unroll
        for (int i = 0; i < 12; ++i) {
            const int e  = tid + (i << 8);            // < 3072 = 64*48
            const int li = e / HDIM;
            const int d  = e - li * HDIM;
            const size_t row = base + (size_t)(lc + li) * 1152;
            k_s[li][d] = qkv[row + CDIM     + hcol + d];
            v_s[li][d] = qkv[row + 2 * CDIM + hcol + d];
        }
        __syncthreads();
        if (tid < 64) {                               // cosine-normalize k rows
            float s = 0.f;
            for (int d = 0; d < HDIM; ++d) { const float v = k_s[tid][d]; s += v * v; }
            const float r = rsqrtf(fmaxf(s, 1e-24f));
            for (int d = 0; d < HDIM; ++d) k_s[tid][d] *= r;
        }
        __syncthreads();
        for (int li = 0; li < 64; ++li) {
            #pragma unroll
            for (int i = 0; i < 9; ++i)
                acc[i] += k_s[li][dd[i]] * v_s[li][ee[i]];
        }
        __syncthreads();
    }
    #pragma unroll
    for (int i = 0; i < 9; ++i) attn_s[dd[i]][ee[i]] = acc[i];
    __syncthreads();

    // ---- pass 2: o = 0.5 v + (1/pi) qn attn ; renorm ; + mid ----
    for (int lc = 0; lc < LTOK; lc += 64) {
        #pragma unroll
        for (int i = 0; i < 12; ++i) {
            const int e  = tid + (i << 8);
            const int li = e / HDIM;
            const int d  = e - li * HDIM;
            const size_t row = base + (size_t)(lc + li) * 1152;
            k_s[li][d] = qkv[row            + hcol + d];   // q
            v_s[li][d] = qkv[row + 2 * CDIM + hcol + d];   // v
        }
        __syncthreads();
        if (tid < 64) {                               // cosine-normalize q rows
            float s = 0.f;
            for (int d = 0; d < HDIM; ++d) { const float v = k_s[tid][d]; s += v * v; }
            const float r = rsqrtf(fmaxf(s, 1e-24f));
            for (int d = 0; d < HDIM; ++d) k_s[tid][d] *= r;
        }
        __syncthreads();
        #pragma unroll
        for (int i = 0; i < 12; ++i) {                // o written in place of v
            const int e  = tid + (i << 8);
            const int li = e / HDIM;
            const int ec = e - li * HDIM;
            float s = 0.f;
            for (int d = 0; d < HDIM; ++d) s += k_s[li][d] * attn_s[d][ec];
            v_s[li][ec] = 0.5f * v_s[li][ec] + INV_PI * s;
        }
        __syncthreads();
        if (tid < 64) {                               // row norms of o
            float s = 0.f;
            for (int ec = 0; ec < HDIM; ++ec) { const float v = v_s[tid][ec]; s += v * v; }
            rnorm[tid] = rsqrtf(fmaxf(s, 1e-24f));
        }
        __syncthreads();
        #pragma unroll
        for (int i = 0; i < 12; ++i) {
            const int e  = tid + (i << 8);
            const int li = e / HDIM;
            const int ec = e - li * HDIM;
            const size_t row = (((size_t)b) << 10) + lc + li;
            const float val = v_s[li][ec] * rnorm[li] + mid[row * CDIM + hcol + ec];
            o[row * CDIM + hcol + ec] = __float2bfloat16(val);
        }
        __syncthreads();
    }
}

// ---------------------------------------------------------------------------
// Host orchestration
// ---------------------------------------------------------------------------
extern "C" void kernel_launch(void* const* d_in, const int* in_sizes, int n_in,
                              void* d_out, int out_size, void* d_ws, size_t ws_size,
                              hipStream_t stream)
{
    (void)in_sizes; (void)n_in; (void)out_size; (void)ws_size;
    const float* x      = (const float*)d_in[0];
    const float* qkv_w  = (const float*)d_in[1];
    const float* proj_w = (const float*)d_in[2];
    const float* proj_b = (const float*)d_in[3];
    const float* ln_g   = (const float*)d_in[4];
    const float* ln_b   = (const float*)d_in[5];
    const float* fc1_w  = (const float*)d_in[6];
    const float* fc1_b  = (const float*)d_in[7];
    const float* fc2_w  = (const float*)d_in[8];
    const float* fc2_b  = (const float*)d_in[9];

    char* ws = (char*)d_ws;
    size_t off = 0;
    auto take = [&](size_t bytes) -> char* {
        char* p = ws + off;
        off = (off + bytes + 255) & ~(size_t)255;
        return p;
    };

    __hip_bfloat16* qkvWt = (__hip_bfloat16*)take((size_t)1152 * CDIM * 2);
    __hip_bfloat16* fc1Wt = (__hip_bfloat16*)take((size_t)HIDDEN * CDIM * 2);
    __hip_bfloat16* fc2Wt = (__hip_bfloat16*)take((size_t)CDIM * HIDDEN * 2);
    __hip_bfloat16* prjWt = (__hip_bfloat16*)take((size_t)CDIM * CDIM * 2);
    __hip_bfloat16* xs    = (__hip_bfloat16*)take((size_t)MROWS * CDIM * 2);
    __hip_bfloat16* xn    = (__hip_bfloat16*)take((size_t)MROWS * CDIM * 2);
    float*          qkvb  = (float*)         take((size_t)MROWS * 1152 * 4);
    __hip_bfloat16* hid   = (__hip_bfloat16*)take((size_t)MROWS * HIDDEN * 2);
    float*          midb  = (float*)         take((size_t)MROWS * CDIM * 4);
    __hip_bfloat16* obuf  = (__hip_bfloat16*)take((size_t)MROWS * CDIM * 2);

    // 1) weight convert/transpose to bf16 [N x K]
    {
        int n;
        n = CDIM * 1152;   convert_transpose_kernel<<<(n + 255) / 256, 256, 0, stream>>>(qkv_w,  qkvWt, CDIM,   1152);
        n = CDIM * HIDDEN; convert_transpose_kernel<<<(n + 255) / 256, 256, 0, stream>>>(fc1_w,  fc1Wt, CDIM,   HIDDEN);
        n = HIDDEN * CDIM; convert_transpose_kernel<<<(n + 255) / 256, 256, 0, stream>>>(fc2_w,  fc2Wt, HIDDEN, CDIM);
        n = CDIM * CDIM;   convert_transpose_kernel<<<(n + 255) / 256, 256, 0, stream>>>(proj_w, prjWt, CDIM,   CDIM);
    }

    // 2) NCHW -> token-major + LayerNorm (bf16 xs, xn)
    ln_kernel<<<1024, 256, 0, stream>>>(x, ln_g, ln_b, xs, xn);

    // 3) qkv = xs @ qkv_w                       [32768 x 1152], f32 out
    wmma_gemm_kernel<0><<<dim3(1152 / 128, MROWS / 128), 256, 0, stream>>>(
        xs, qkvWt, nullptr, qkvb, MROWS, 1152, CDIM);

    // 4) hid = silu(xn @ fc1_w + b)             [32768 x 1536], bf16 out
    wmma_gemm_kernel<1><<<dim3(HIDDEN / 128, MROWS / 128), 256, 0, stream>>>(
        xn, fc1Wt, fc1_b, hid, MROWS, HIDDEN, CDIM);

    // 5) mid = hid @ fc2_w + b                  [32768 x 384], f32 out
    wmma_gemm_kernel<2><<<dim3(CDIM / 128, MROWS / 128), 256, 0, stream>>>(
        hid, fc2Wt, fc2_b, midb, MROWS, CDIM, HIDDEN);

    // 6) linear attention + renorm + mid, bf16 [32768 x 384]
    attn_kernel<<<BATCH * NHEAD, 256, 0, stream>>>(qkvb, midb, obuf);

    // 7) out = o @ proj_w + b, scattered to [B, C, H, W] f32
    wmma_gemm_kernel<3><<<dim3(CDIM / 128, MROWS / 128), 256, 0, stream>>>(
        obuf, prjWt, proj_b, (float*)d_out, MROWS, CDIM, CDIM);
}
